// FoundationGNN_84567906058443
// MI455X (gfx1250) — matrix operand order repaired and checked by reading, not compiled, not run on hardware
//
#include <hip/hip_runtime.h>
#include <stdint.h>

#define IN_DIM  15
#define HID     128
#define HEADS   8
#define LAYERS  6
#define NNODES  20000
#define NEDGES  160000
#define HC      1024   // HEADS*HID

typedef __bf16 v16bf __attribute__((ext_vector_type(16)));
typedef float  v8f   __attribute__((ext_vector_type(8)));
typedef unsigned int v4u __attribute__((ext_vector_type(4)));
typedef int v4i __attribute__((ext_vector_type(4)));
typedef int v8i __attribute__((ext_vector_type(8)));

// ---------- helpers ----------
__device__ __forceinline__ uint16_t f2bfu(float f) {
  union { float f; uint32_t u; } v; v.f = f;
  uint32_t r = v.u + 0x7fffu + ((v.u >> 16) & 1u);   // round-to-nearest-even
  return (uint16_t)(r >> 16);
}

__device__ __forceinline__ void atomicMaxF(float* addr, float val) {
  if (val >= 0.0f) atomicMax((int*)addr, __float_as_int(val));
  else             atomicMin((unsigned int*)addr, __float_as_uint(val));
}

__device__ __forceinline__ float blockReduce256(float v, float* red, int tid) {
  #pragma unroll
  for (int o = 16; o > 0; o >>= 1) v += __shfl_xor(v, o, 32);
  __syncthreads();
  if ((tid & 31) == 0) red[tid >> 5] = v;
  __syncthreads();
  if (tid == 0) { float s = 0.f; for (int i = 0; i < 8; ++i) s += red[i]; red[8] = s; }
  __syncthreads();
  return red[8];
}

// ---------- Tensor Data Mover: 1-D contiguous tile global -> LDS ----------
// D# per CDNA5 ISA: group0 = {count, lds_addr, global_addr, type=2},
// group1 = {data_size=8B, tensor_dim0=n8, tensor_dim1=1, tile_dim0=n8, tile_dim1=1, stride=n8}
__device__ __forceinline__ void tdm_load_1d(uint32_t lds_off, const void* gptr, uint32_t bytes) {
  const uint64_t ga = (uint64_t)(uintptr_t)gptr;
  const uint32_t n8 = bytes >> 3;                   // 8-byte units (tile_dim0 is 16-bit)
  v4u g0;
  g0.x = 1u;                                        // count=1, user descriptor
  g0.y = lds_off;                                   // LDS byte address
  g0.z = (uint32_t)ga;                              // global_addr[31:0]
  g0.w = (uint32_t)((ga >> 32) & 0x1ffffffu) | (2u << 30);  // global_addr[56:32] | type=2
  v8i g1;
  g1[0] = (int)(3u << 16);                          // data_size=3 (8 bytes)
  g1[1] = (int)((n8 & 0xffffu) << 16);              // tensor_dim0[15:0]
  g1[2] = (int)(((n8 >> 16) & 0xffffu) | (1u << 16)); // tensor_dim0[31:16] | tensor_dim1=1
  g1[3] = (int)((n8 & 0xffffu) << 16);              // tile_dim0 = n8
  g1[4] = 1;                                        // tile_dim1=1, tile_dim2=0
  g1[5] = (int)n8;                                  // tensor_dim0_stride[31:0]
  g1[6] = 0;
  g1[7] = 0;
  const v4i z4 = {0, 0, 0, 0};
#if __has_include(<hip/amd_detail/amd_gfx1250_TDM.h>)
  const v8i z8 = {0, 0, 0, 0, 0, 0, 0, 0};
  __builtin_amdgcn_tensor_load_to_lds(g0, g1, z4, z4, z8, 0);
#else
  __builtin_amdgcn_tensor_load_to_lds(g0, g1, z4, z4, 0);
#endif
}

// ---------- fill ----------
__global__ void k_fill(float* __restrict__ p, size_t n, float val) {
  size_t i = (size_t)blockIdx.x * blockDim.x + threadIdx.x;
  if (i < n) p[i] = val;
}

// ---------- input projection: h = x @ Win + b_in ----------
__global__ __launch_bounds__(128) void k_input_proj(
    const float* __restrict__ x, const float* __restrict__ Win,
    const float* __restrict__ b_in, float* __restrict__ h) {
  __shared__ float xs[IN_DIM];
  const int n = blockIdx.x, tid = threadIdx.x;
  if (tid < IN_DIM) xs[tid] = x[(size_t)n * IN_DIM + tid];
  __syncthreads();
  float s = b_in[tid];
  #pragma unroll
  for (int i = 0; i < IN_DIM; ++i) s += xs[i] * Win[i * HID + tid];
  h[(size_t)n * HID + tid] = s;
}

// ---------- swizzle A [M x K] f32 -> WMMA A fragments (bf16) ----------
// tile order: tileid = mt*KT + kt ; each tile = 32 lanes x 16 bf16 (32B/lane)
// lanes 0-15: M=lane,    K = 0..15 ; lanes 16-31: M=lane-16, K = 16..31
__global__ __launch_bounds__(256) void k_swizzleA(
    const float* __restrict__ A, int K, int KT, __bf16* __restrict__ out, int total) {
  int t = blockIdx.x * blockDim.x + threadIdx.x;
  if (t >= total) return;
  const int jhalf  = t & 1;
  const int lane   = (t >> 1) & 31;
  const int tileid = t >> 6;
  const int kt = tileid % KT;
  const int mt = tileid / KT;
  const int hs = lane >> 4, lm = lane & 15;
  const int row   = mt * 16 + lm;
  const int kbase = kt * 32 + hs * 16 + jhalf * 8;
  const float* src = A + (size_t)row * K + kbase;
  uint16_t u[8];
  #pragma unroll
  for (int j = 0; j < 8; ++j) u[j] = f2bfu(src[j]);
  uint4 w;
  w.x = (uint32_t)u[0] | ((uint32_t)u[1] << 16);
  w.y = (uint32_t)u[2] | ((uint32_t)u[3] << 16);
  w.z = (uint32_t)u[4] | ((uint32_t)u[5] << 16);
  w.w = (uint32_t)u[6] | ((uint32_t)u[7] << 16);
  *(uint4*)(out + ((size_t)tileid * 32 + lane) * 16 + jhalf * 8) = w;
}

// ---------- swizzle B [K x N] f32 -> WMMA B fragments (bf16), TDM-panel-major ----------
// panel = 4 consecutive n-tiles (one GEMM block). Output tile order:
//   otile = ((nt>>2)*KT + kt)*4 + (nt&3)   -> each block's panel is contiguous (KT*4 tiles)
// within tile: lanes 0-15: N=lane, K=0..15 ; lanes 16-31: N=lane-16, K=16..31
__global__ __launch_bounds__(256) void k_swizzleB(
    const float* __restrict__ W, int N, int NT, int KT,
    __bf16* __restrict__ out, int total) {
  int t = blockIdx.x * blockDim.x + threadIdx.x;
  if (t >= total) return;
  const int jhalf  = t & 1;
  const int lane   = (t >> 1) & 31;
  const int tileid = t >> 6;
  const int nt = tileid % NT;
  const int kt = tileid / NT;
  const int hs = lane >> 4, ln = lane & 15;
  const int n     = nt * 16 + ln;
  const int kbase = kt * 32 + hs * 16 + jhalf * 8;
  uint16_t u[8];
  #pragma unroll
  for (int j = 0; j < 8; ++j) u[j] = f2bfu(W[(size_t)(kbase + j) * N + n]);
  uint4 w;
  w.x = (uint32_t)u[0] | ((uint32_t)u[1] << 16);
  w.y = (uint32_t)u[2] | ((uint32_t)u[3] << 16);
  w.z = (uint32_t)u[4] | ((uint32_t)u[5] << 16);
  w.w = (uint32_t)u[6] | ((uint32_t)u[7] << 16);
  const int otile = ((nt >> 2) * KT + kt) * 4 + (nt & 3);
  *(uint4*)(out + ((size_t)otile * 32 + lane) * 16 + jhalf * 8) = w;
}

// ---------- WMMA GEMM: C[M x N] = A @ B (+bias) (+resid, relu) ----------
// B panel staged in LDS by the Tensor Data Mover in chunks of up to 8 k-tiles (32KB).
// EPI 0: C = acc + bias ; EPI 1: C = relu(acc + bias + resid)
template <int EPI>
__global__ __launch_bounds__(256) void k_gemm_wmma(
    const __bf16* __restrict__ Asw, const __bf16* __restrict__ Bsw,
    const float* __restrict__ bias, const float* __restrict__ resid,
    float* __restrict__ C, int MT, int KT, int N) {
  __shared__ __align__(32) __bf16 ldsB[8 * 4 * 512];   // 32 KB panel chunk
  const int wave = threadIdx.x >> 5;
  const int lane = threadIdx.x & 31;
  const int mt = blockIdx.y * 8 + wave;
  const int mtc = (mt < MT) ? mt : (MT - 1);           // clamp; barriers need all waves

  const v16bf* Ap = (const v16bf*)Asw + (size_t)mtc * KT * 32 + lane;
  const __bf16* panel = Bsw + (size_t)blockIdx.x * KT * 2048;  // 4 tiles * 512 elems per kt
  const v16bf* Bl = (const v16bf*)ldsB;
  const uint32_t ldsOff = (uint32_t)(uintptr_t)(void*)ldsB;

  v8f acc[4];
  const v8f vz = {0.f, 0.f, 0.f, 0.f, 0.f, 0.f, 0.f, 0.f};
  #pragma unroll
  for (int t = 0; t < 4; ++t) acc[t] = vz;

  for (int kc = 0; kc < KT; kc += 8) {
    const int ch = (KT - kc < 8) ? (KT - kc) : 8;
    __syncthreads();                                   // prior chunk consumed
    if (wave == 0) {
      tdm_load_1d(ldsOff, panel + (size_t)kc * 2048, (uint32_t)ch * 4096u);
      __builtin_amdgcn_s_wait_tensorcnt(0);
    }
    __syncthreads();                                   // panel chunk visible
    for (int kt2 = 0; kt2 < ch; ++kt2) {
      v16bf a = Ap[(size_t)(kc + kt2) * 32];
      if (kt2 + 1 < ch)
        __builtin_prefetch((const void*)(Ap + (size_t)(kc + kt2 + 1) * 32), 0, 0);
      #pragma unroll
      for (int t = 0; t < 4; ++t) {
        v16bf b = Bl[(size_t)(kt2 * 4 + t) * 32 + lane];
        acc[t] = __builtin_amdgcn_wmma_f32_16x16x32_bf16(
            false, a, false, b, (short)0, acc[t], false, false);
      }
    }
  }

  if (mt >= MT) return;
  // C/D layout: VGPR i -> row (hs*8 + i), col = lane&15 within tile
  const int hs = lane >> 4, cl = lane & 15;
  const int rbase = mt * 16 + hs * 8;
  const int nt0 = blockIdx.x * 4;
  #pragma unroll
  for (int t = 0; t < 4; ++t) {
    const int col = (nt0 + t) * 16 + cl;
    const float bv = bias[col];
    #pragma unroll
    for (int i = 0; i < 8; ++i) {
      const int r = rbase + i;
      float xv = acc[t][i] + bv;
      if (EPI == 1) { xv += resid[(size_t)r * N + col]; xv = fmaxf(xv, 0.f); }
      C[(size_t)r * N + col] = xv;
    }
  }
}

// ---------- edge attention logits + running segment max ----------
__global__ __launch_bounds__(256) void k_logits(
    const float* __restrict__ q, const float* __restrict__ kk,
    const int* __restrict__ src, const int* __restrict__ dst,
    float* __restrict__ alpha, float* __restrict__ amax) {
  const int wave = threadIdx.x >> 5, lane = threadIdx.x & 31;
  const int e = blockIdx.x * 8 + wave;
  if (e >= NEDGES) return;
  const int s = src[e], d = dst[e];
  const int hg = lane >> 2, sub = lane & 3;
  const float4* qp = (const float4*)(q  + (size_t)d * HC + hg * HID + sub * 32);
  const float4* kp = (const float4*)(kk + (size_t)s * HC + hg * HID + sub * 32);
  float acc = 0.f;
  #pragma unroll
  for (int i = 0; i < 8; ++i) {
    float4 a = qp[i], b = kp[i];
    acc += a.x * b.x + a.y * b.y + a.z * b.z + a.w * b.w;
  }
  acc += __shfl_xor(acc, 1, 32);
  acc += __shfl_xor(acc, 2, 32);
  if (sub == 0) {
    const float al = acc * 0.08838834764831845f;  // 1/sqrt(128)
    alpha[(size_t)e * HEADS + hg] = al;
    atomicMaxF(&amax[(size_t)d * HEADS + hg], al);
  }
}

// ---------- exp(alpha - amax[dst]) and segment sum ----------
__global__ __launch_bounds__(256) void k_expsum(
    float* __restrict__ alpha, const float* __restrict__ amax,
    const int* __restrict__ dst, float* __restrict__ denom) {
  const int i = blockIdx.x * blockDim.x + threadIdx.x;
  if (i >= NEDGES * HEADS) return;
  const int e = i >> 3, hh = i & 7;
  const int d = dst[e];
  const float ex = __expf(alpha[i] - amax[(size_t)d * HEADS + hh]);
  alpha[i] = ex;
  atomicAdd(&denom[(size_t)d * HEADS + hh], ex);
}

// ---------- weighted message scatter: agg[dst] += attn * v[src] ----------
__global__ __launch_bounds__(256) void k_scatter(
    const float* __restrict__ v, const float* __restrict__ ex,
    const float* __restrict__ denom, const int* __restrict__ src,
    const int* __restrict__ dst, float* __restrict__ agg) {
  const int e = blockIdx.x, tid = threadIdx.x;
  const int s = src[e], d = dst[e];
  const int hh = tid >> 5;  // (tid*4)/128
  const float w = ex[(size_t)e * HEADS + hh] /
                  (denom[(size_t)d * HEADS + hh] + 1e-16f);
  const float4 vv = *(const float4*)(v + (size_t)s * HC + tid * 4);
  float* op = agg + (size_t)d * HC + tid * 4;
  atomicAdd(op + 0, w * vv.x);
  atomicAdd(op + 1, w * vv.y);
  atomicAdd(op + 2, w * vv.z);
  atomicAdd(op + 3, w * vv.w);
}

// ---------- beta-gated skip + LayerNorm (in place on agg) ----------
__global__ __launch_bounds__(256) void k_beta_ln(
    float* __restrict__ agg, const float* __restrict__ xr,
    const float* __restrict__ Wb, const float* __restrict__ g,
    const float* __restrict__ bb) {
  __shared__ float red[16];
  const int n = blockIdx.x, tid = threadIdx.x;
  float o[4], r[4];
  #pragma unroll
  for (int i = 0; i < 4; ++i) {
    const int idx = tid + i * 256;
    o[i] = agg[(size_t)n * HC + idx];
    r[i] = xr [(size_t)n * HC + idx];
  }
  float s = 0.f;
  #pragma unroll
  for (int i = 0; i < 4; ++i) {
    const int idx = tid + i * 256;
    s += o[i] * Wb[idx] + r[i] * Wb[HC + idx] + (o[i] - r[i]) * Wb[2 * HC + idx];
  }
  s = blockReduce256(s, red, tid);
  const float beta = 1.f / (1.f + __expf(-s));
  float c[4], sum = 0.f, sq = 0.f;
  #pragma unroll
  for (int i = 0; i < 4; ++i) {
    c[i] = beta * r[i] + (1.f - beta) * o[i];
    sum += c[i]; sq += c[i] * c[i];
  }
  sum = blockReduce256(sum, red, tid);
  sq  = blockReduce256(sq,  red, tid);
  const float mu  = sum * (1.f / HC);
  const float var = sq * (1.f / HC) - mu * mu;
  const float rs  = rsqrtf(var + 1e-5f);
  #pragma unroll
  for (int i = 0; i < 4; ++i) {
    const int idx = tid + i * 256;
    agg[(size_t)n * HC + idx] = (c[i] - mu) * rs * g[idx] + bb[idx];
  }
}

// ---------- host orchestration ----------
extern "C" void kernel_launch(void* const* d_in, const int* in_sizes, int n_in,
                              void* d_out, int out_size, void* d_ws, size_t ws_size,
                              hipStream_t stream) {
  const float* x     = (const float*)d_in[0];
  const int*   ei    = (const int*)  d_in[1];
  const float* Win   = (const float*)d_in[2];
  const float* b_in  = (const float*)d_in[3];
  const float* Wq    = (const float*)d_in[4];
  const float* bq    = (const float*)d_in[5];
  const float* Wk    = (const float*)d_in[6];
  const float* bk    = (const float*)d_in[7];
  const float* Wv    = (const float*)d_in[8];
  const float* bv    = (const float*)d_in[9];
  const float* Wsk   = (const float*)d_in[10];
  const float* bsk   = (const float*)d_in[11];
  const float* Wbeta = (const float*)d_in[12];
  const float* ln_g  = (const float*)d_in[13];
  const float* ln_b  = (const float*)d_in[14];
  const float* Wproj = (const float*)d_in[15];
  const float* bproj = (const float*)d_in[16];
  const int* src = ei;
  const int* dst = ei + NEDGES;

  // workspace carve-out
  char* p = (char*)d_ws;
  auto alloc = [&](size_t bytes) -> char* {
    char* r = p; p += (bytes + 255) & ~(size_t)255; return r;
  };
  float*  hA    = (float*)alloc((size_t)NNODES * HID * 4);
  float*  hB    = (float*)alloc((size_t)NNODES * HID * 4);
  float*  qb    = (float*)alloc((size_t)NNODES * HC * 4);
  float*  kb    = (float*)alloc((size_t)NNODES * HC * 4);
  float*  vb    = (float*)alloc((size_t)NNODES * HC * 4);
  float*  xrb   = (float*)alloc((size_t)NNODES * HC * 4);
  float*  agg   = (float*)alloc((size_t)NNODES * HC * 4);
  float*  alpha = (float*)alloc((size_t)NEDGES * HEADS * 4);
  float*  amax  = (float*)alloc((size_t)NNODES * HEADS * 4);
  float*  denom = (float*)alloc((size_t)NNODES * HEADS * 4);
  __bf16* hsw   = (__bf16*)alloc((size_t)NNODES * HID * 2);
  __bf16* aggsw = (__bf16*)alloc((size_t)NNODES * HC * 2);
  __bf16* wsw   = (__bf16*)alloc((size_t)5 * LAYERS * HID * HC * 2);
  if ((size_t)(p - (char*)d_ws) > ws_size) return;  // not enough scratch

  const int MT = NNODES / 16;         // 1250
  const int WELEMS = HID * HC;        // 131072 floats per weight per layer
  auto wswAt = [&](int w, int l) { return wsw + (size_t)(w * LAYERS + l) * WELEMS; };

  // swizzle all weights to TDM-panel-major WMMA-B fragment layout (bf16)
  {
    const int nthr = HID * HC / 8;    // 16384 threads per weight
    dim3 g((nthr + 255) / 256), b(256);
    for (int l = 0; l < LAYERS; ++l) {
      k_swizzleB<<<g, b, 0, stream>>>(Wq    + (size_t)l * WELEMS, HC,  HC / 16,  4,  wswAt(0, l), nthr);
      k_swizzleB<<<g, b, 0, stream>>>(Wk    + (size_t)l * WELEMS, HC,  HC / 16,  4,  wswAt(1, l), nthr);
      k_swizzleB<<<g, b, 0, stream>>>(Wv    + (size_t)l * WELEMS, HC,  HC / 16,  4,  wswAt(2, l), nthr);
      k_swizzleB<<<g, b, 0, stream>>>(Wsk   + (size_t)l * WELEMS, HC,  HC / 16,  4,  wswAt(3, l), nthr);
      k_swizzleB<<<g, b, 0, stream>>>(Wproj + (size_t)l * WELEMS, HID, HID / 16, 32, wswAt(4, l), nthr);
    }
  }

  // h0 = x @ Win + b_in
  k_input_proj<<<NNODES, 128, 0, stream>>>(x, Win, b_in, hA);

  float* h_cur = hA;
  float* h_nxt = hB;
  const dim3 gemmGridQ(HC / 64, (MT + 7) / 8);    // 16 x 157
  const dim3 gemmGridP(HID / 64, (MT + 7) / 8);   // 2 x 157

  for (int l = 0; l < LAYERS; ++l) {
    // A-swizzle of h (K=128, KT=4)
    {
      const int tot = MT * 4 * 64;  // 320000
      k_swizzleA<<<(tot + 255) / 256, 256, 0, stream>>>(h_cur, HID, 4, hsw, tot);
    }
    // q/k/v/skip GEMMs
    k_gemm_wmma<0><<<gemmGridQ, 256, 0, stream>>>(hsw, wswAt(0, l), bq  + (size_t)l * HC, nullptr, qb,  MT, 4, HC);
    k_gemm_wmma<0><<<gemmGridQ, 256, 0, stream>>>(hsw, wswAt(1, l), bk  + (size_t)l * HC, nullptr, kb,  MT, 4, HC);
    k_gemm_wmma<0><<<gemmGridQ, 256, 0, stream>>>(hsw, wswAt(2, l), bv  + (size_t)l * HC, nullptr, vb,  MT, 4, HC);
    k_gemm_wmma<0><<<gemmGridQ, 256, 0, stream>>>(hsw, wswAt(3, l), bsk + (size_t)l * HC, nullptr, xrb, MT, 4, HC);

    // segment-softmax scratch init
    k_fill<<<((size_t)NNODES * HC + 255) / 256, 256, 0, stream>>>(agg, (size_t)NNODES * HC, 0.f);
    k_fill<<<((size_t)NNODES * HEADS + 255) / 256, 256, 0, stream>>>(amax, (size_t)NNODES * HEADS, -3.402823466e38f);
    k_fill<<<((size_t)NNODES * HEADS + 255) / 256, 256, 0, stream>>>(denom, (size_t)NNODES * HEADS, 0.f);

    // attention
    k_logits<<<NEDGES / 8, 256, 0, stream>>>(qb, kb, src, dst, alpha, amax);
    k_expsum<<<(NEDGES * HEADS + 255) / 256, 256, 0, stream>>>(alpha, amax, dst, denom);
    k_scatter<<<NEDGES, 256, 0, stream>>>(vb, alpha, denom, src, dst, agg);

    // beta-gated skip + LayerNorm (in place)
    k_beta_ln<<<NNODES, 256, 0, stream>>>(agg, xrb,
                                          Wbeta + (size_t)l * 3 * HC,
                                          ln_g + (size_t)l * HC,
                                          ln_b + (size_t)l * HC);

    // A-swizzle of LN output (K=1024, KT=32), then proj GEMM with residual+relu
    {
      const int tot = MT * 32 * 64;  // 2,560,000
      k_swizzleA<<<(tot + 255) / 256, 256, 0, stream>>>(agg, HC, 32, aggsw, tot);
    }
    float* outp = (l == LAYERS - 1) ? (float*)d_out : h_nxt;
    k_gemm_wmma<1><<<gemmGridP, 256, 0, stream>>>(aggsw, wswAt(4, l),
                                                  bproj + (size_t)l * HID, h_cur,
                                                  outp, MT, 32, HID);
    // ping-pong
    float* t = h_cur; h_cur = h_nxt; h_nxt = t;
    if (l == LAYERS - 1) h_cur = (float*)d_out;
  }
}
